// ResidualBlock_77498389889548
// MI455X (gfx1250) — matrix operand order repaired and checked by reading, not compiled, not run on hardware
//
#include <hip/hip_runtime.h>

typedef __attribute__((ext_vector_type(16))) _Float16 v16h;
typedef __attribute__((ext_vector_type(8)))  float    v8f;
typedef __attribute__((ext_vector_type(4)))  unsigned int u32x4;
typedef __attribute__((ext_vector_type(8)))  int      i32x8;
typedef __attribute__((ext_vector_type(4)))  int      i32x4;

#define NPTS 32768
#define KOFF 27
#define CH   128
#define EPSV 1e-5f

#if defined(__has_builtin)
#if __has_builtin(__builtin_amdgcn_tensor_load_to_lds) && \
    __has_builtin(__builtin_amdgcn_s_wait_tensorcnt)
#define HAVE_TDM 1
#endif
#endif

union HF16 { v16h h; uint4 q[2]; };

// ---------------------------------------------------------------------------
// Prep: f32 -> f16 elementwise
__global__ void f32_to_f16_kernel(const float* __restrict__ x,
                                  _Float16* __restrict__ y, int n) {
    int i = blockIdx.x * blockDim.x + threadIdx.x;
    if (i < n) y[i] = (_Float16)x[i];
}

__global__ void zero_f32_kernel(float* __restrict__ p, int n) {
    int i = blockIdx.x * blockDim.x + threadIdx.x;
    if (i < n) p[i] = 0.0f;
}

// ---------------------------------------------------------------------------
// Pack W [K][C][C] f32 (row = input channel c, col = output channel d) into
// per-fragment B layout for v_wmma_f32_16x16x32_f16:
//   fragment (k, cc, t): 32 lanes x 16 halves, contiguous per lane (32B).
//   lane L, half j:  K_local = (L>>4)*16 + j  (c = cc*32 + K_local)
//                    column  = t*16 + (L&15)  (d)
__global__ void pack_w_kernel(const float* __restrict__ W,
                              _Float16* __restrict__ Wp) {
    int idx = blockIdx.x * blockDim.x + threadIdx.x;
    if (idx >= KOFF * CH * CH) return;
    int j    = idx & 15;
    int lane = (idx >> 4) & 31;
    int t    = (idx >> 9) & 7;
    int cc   = (idx >> 12) & 3;
    int k    = idx >> 14;
    int c    = cc * 32 + (lane >> 4) * 16 + j;
    int d    = t * 16 + (lane & 15);
    Wp[idx] = (_Float16)W[(k * CH + c) * CH + d];
}

// ---------------------------------------------------------------------------
#ifdef HAVE_TDM
// Issue TDM: contiguous 32KB copy (4096 x 8-byte elements) global -> LDS.
__device__ __forceinline__ void tdm_load_slice(const void* gptr, unsigned lds_off) {
    unsigned long long ga = (unsigned long long)gptr;
    u32x4 g0;
    g0.x = 1u;                                   // count=1, user descriptor
    g0.y = lds_off;                              // lds_addr (bytes)
    g0.z = (unsigned)(ga & 0xffffffffu);         // global_addr[31:0]
    g0.w = (unsigned)((ga >> 32) & 0x1ffffffu)   // global_addr[56:32]
         | 0x80000000u;                          // type=2 ("image") bits[127:126]
    i32x8 g1;
    g1[0] = 0x00030000;                          // data_size=3 (8B), mask=0
    g1[1] = (int)(4096u << 16);                  // tensor_dim0[15:0] in [31:16]
    g1[2] = (int)(1u << 16);                     // tensor_dim0 hi=0 ; tensor_dim1=1
    g1[3] = (int)(4096u << 16);                  // tensor_dim1 hi=0 ; tile_dim0=4096
    g1[4] = 0;                                   // tile_dim1=0 (unused), tile_dim2=0
    g1[5] = 4096;                                // tensor_dim0_stride[31:0]
    g1[6] = 0;                                   // stride hi / dim1_stride lo
    g1[7] = 0;
    i32x4 gz = {0, 0, 0, 0};
#if defined(__clang_major__) && (__clang_major__ >= 23)
    i32x8 gz8 = {0, 0, 0, 0, 0, 0, 0, 0};
    __builtin_amdgcn_tensor_load_to_lds(g0, g1, gz, gz, gz8, 0);
#else
    __builtin_amdgcn_tensor_load_to_lds(g0, g1, gz, gz, 0);
#endif
}
#endif

// ---------------------------------------------------------------------------
// Sparse-conv GEMM: out[n,d] = sum_k sum_c feats16[nbr[n,k], c] * W[k][c,d]
// 256-thread block = 8 waves, each wave owns a 16-row M tile (128 rows/block).
// W[k] slice (32KB) staged in LDS (TDM double buffer), shared by all 8 waves.
__global__ __launch_bounds__(256)
void conv_wmma_kernel(const _Float16* __restrict__ feats16,
                      const int*      __restrict__ nbr,
                      const _Float16* __restrict__ Wp,
                      float*          __restrict__ out) {
    __shared__ uint4 smem[2][2048];              // 2 x 32KB W-slice buffers

    const int lane = threadIdx.x & 31;
    const int wave = threadIdx.x >> 5;
    const int tile = blockIdx.x * 8 + wave;      // uniform per wave
    const int n0   = tile * 16;
    const int m    = lane & 15;
    const int hi   = lane >> 4;
    const int n    = n0 + m;                     // this lane's A-matrix row

    const v8f vzero = {0.f, 0.f, 0.f, 0.f, 0.f, 0.f, 0.f, 0.f};
    v8f acc[8];
#pragma unroll
    for (int t = 0; t < 8; ++t) acc[t] = vzero;

#ifdef HAVE_TDM
    // Prologue: prefetch slice k=0 into buffer 0 (one TDM issue per block).
    if (wave == 0) {
        tdm_load_slice(Wp, (unsigned)(unsigned long long)&smem[0][0]);
    }
#endif

    for (int k = 0; k < KOFF; ++k) {
        const int buf = k & 1;
#ifdef HAVE_TDM
        if (wave == 0) {
            __builtin_amdgcn_s_wait_tensorcnt(0);   // slice k resident in LDS
        }
        __syncthreads();                            // publish LDS, retire k-1 use
        if (wave == 0 && k + 1 < KOFF) {            // prefetch slice k+1
            tdm_load_slice(Wp + (size_t)(k + 1) * (CH * CH),
                           (unsigned)(unsigned long long)&smem[buf ^ 1][0]);
        }
#else
        // Fallback: cooperative global->LDS copy of slice k.
        __syncthreads();
        {
            const uint4* src = (const uint4*)(Wp + (size_t)k * (CH * CH));
            for (int i = threadIdx.x; i < 2048; i += 256) smem[buf][i] = src[i];
        }
        __syncthreads();
#endif
        const int nb = nbr[n * KOFF + k];
        const _Float16* arow = feats16 + (size_t)nb * CH;
#pragma unroll
        for (int cc = 0; cc < 4; ++cc) {
            // A fragment 16x32 f16: halves 0..15 = K hi*16+0..15
            // -> two contiguous 16B loads.
            HF16 a;
            a.q[0] = *(const uint4*)(arow + cc * 32 + hi * 16);
            a.q[1] = *(const uint4*)(arow + cc * 32 + hi * 16 + 8);
#pragma unroll
            for (int t = 0; t < 8; ++t) {
                // B fragment from LDS: 32 lanes x 32B, ds_load_b128 x2.
                const int fo = ((cc * 8 + t) * 32 + lane) * 2;
                HF16 b;
                b.q[0] = smem[buf][fo];
                b.q[1] = smem[buf][fo + 1];
                acc[t] = __builtin_amdgcn_wmma_f32_16x16x32_f16(
                    /*neg_a=*/false, a.h, /*neg_b=*/false, b.h,
                    /*c_mod=*/(short)0, acc[t],
                    /*reuse_a=*/false, /*reuse_b=*/false);
            }
        }
    }

    // C/D layout: acc[t] element r -> row = n0 + r + 8*hi, col = t*16 + m
#pragma unroll
    for (int r = 0; r < 8; ++r) {
        float* orow = out + (size_t)(n0 + r + 8 * hi) * CH + m;
#pragma unroll
        for (int t = 0; t < 8; ++t) orow[t * 16] = acc[t][r];
    }
}

// ---------------------------------------------------------------------------
// Per-channel sum / sumsq over a 128-row chunk (coalesced), atomic into stats.
__global__ __launch_bounds__(128)
void col_stats_kernel(const float* __restrict__ x,
                      float* __restrict__ sum, float* __restrict__ sumsq) {
    const int d  = threadIdx.x;
    const int r0 = blockIdx.x * 128;
    float s = 0.f, q = 0.f;
    for (int r = 0; r < 128; ++r) {
        float v = x[(size_t)(r0 + r) * CH + d];
        s += v;
        q += v * v;
    }
    atomicAdd(&sum[d], s);
    atomicAdd(&sumsq[d], q);
}

__global__ void bn_finalize_kernel(const float* __restrict__ sum,
                                   const float* __restrict__ sumsq,
                                   const float* __restrict__ g,
                                   const float* __restrict__ b,
                                   float* __restrict__ scale,
                                   float* __restrict__ shift) {
    int d = threadIdx.x;
    float mu  = sum[d] * (1.0f / NPTS);
    float var = sumsq[d] * (1.0f / NPTS) - mu * mu;
    float sc  = g[d] * rsqrtf(var + EPSV);
    scale[d]  = sc;
    shift[d]  = b[d] - mu * sc;
}

// BN + ReLU, emit f16 feature table for next conv.
__global__ void bn_relu_f16_kernel(const float* __restrict__ x,
                                   const float* __restrict__ scale,
                                   const float* __restrict__ shift,
                                   _Float16* __restrict__ y) {
    int i = blockIdx.x * blockDim.x + threadIdx.x;
    int d = i & (CH - 1);
    float v = fmaxf(x[i] * scale[d] + shift[d], 0.0f);
    y[i] = (_Float16)v;
}

// BN + residual + ReLU, f32 output.
__global__ void bn_res_relu_kernel(const float* __restrict__ x,
                                   const float* __restrict__ scale,
                                   const float* __restrict__ shift,
                                   const float* __restrict__ feats,
                                   float* __restrict__ out) {
    int i = blockIdx.x * blockDim.x + threadIdx.x;
    int d = i & (CH - 1);
    float v = x[i] * scale[d] + shift[d] + feats[i];
    out[i] = fmaxf(v, 0.0f);
}

// ---------------------------------------------------------------------------
extern "C" void kernel_launch(void* const* d_in, const int* in_sizes, int n_in,
                              void* d_out, int out_size, void* d_ws, size_t ws_size,
                              hipStream_t stream) {
    const float* feats = (const float*)d_in[0];
    const int*   nbr   = (const int*)d_in[1];
    const float* W1    = (const float*)d_in[2];
    const float* g1    = (const float*)d_in[3];
    const float* b1    = (const float*)d_in[4];
    const float* W2    = (const float*)d_in[5];
    const float* g2    = (const float*)d_in[6];
    const float* b2    = (const float*)d_in[7];
    float* out = (float*)d_out;

    char* ws = (char*)d_ws;
    size_t off = 0;
    auto alloc = [&](size_t bytes) -> void* {
        void* p = ws + off;
        off += (bytes + 255) & ~(size_t)255;
        return p;
    };
    _Float16* feats16 = (_Float16*)alloc((size_t)NPTS * CH * 2);
    _Float16* mid16   = (_Float16*)alloc((size_t)NPTS * CH * 2);
    _Float16* W1p     = (_Float16*)alloc((size_t)KOFF * CH * CH * 2);
    _Float16* W2p     = (_Float16*)alloc((size_t)KOFF * CH * CH * 2);
    float*    convout = (float*)alloc((size_t)NPTS * CH * 4);
    float*    stats   = (float*)alloc((size_t)8 * CH * 4);

    float* sum1 = stats + 0 * CH; float* sq1 = stats + 1 * CH;
    float* sc1  = stats + 2 * CH; float* sh1 = stats + 3 * CH;
    float* sum2 = stats + 4 * CH; float* sq2 = stats + 5 * CH;
    float* sc2  = stats + 6 * CH; float* sh2 = stats + 7 * CH;

    const int NE = NPTS * CH;            // 4194304
    const int WE = KOFF * CH * CH;       // 442368

    zero_f32_kernel<<<(8 * CH + 255) / 256, 256, 0, stream>>>(stats, 8 * CH);
    f32_to_f16_kernel<<<NE / 256, 256, 0, stream>>>(feats, feats16, NE);
    pack_w_kernel<<<(WE + 255) / 256, 256, 0, stream>>>(W1, W1p);
    pack_w_kernel<<<(WE + 255) / 256, 256, 0, stream>>>(W2, W2p);

    // conv1 -> convout
    conv_wmma_kernel<<<NPTS / 128, 256, 0, stream>>>(feats16, nbr, W1p, convout);
    col_stats_kernel<<<NPTS / 128, 128, 0, stream>>>(convout, sum1, sq1);
    bn_finalize_kernel<<<1, 128, 0, stream>>>(sum1, sq1, g1, b1, sc1, sh1);
    bn_relu_f16_kernel<<<NE / 256, 256, 0, stream>>>(convout, sc1, sh1, mid16);

    // conv2 -> convout (buffer reuse)
    conv_wmma_kernel<<<NPTS / 128, 256, 0, stream>>>(mid16, nbr, W2p, convout);
    col_stats_kernel<<<NPTS / 128, 128, 0, stream>>>(convout, sum2, sq2);
    bn_finalize_kernel<<<1, 128, 0, stream>>>(sum2, sq2, g2, b2, sc2, sh2);
    bn_res_relu_kernel<<<NE / 256, 256, 0, stream>>>(convout, sc2, sh2, feats, out);
}
